// TransformerBlock_13881334301431
// MI455X (gfx1250) — matrix operand order repaired and checked by reading, not compiled, not run on hardware
//
#include <hip/hip_runtime.h>

#define NTOK  4096
#define CDIM  768
#define NHEAD 12
#define HDIM  64
#define HID   3072

typedef __bf16 bf16_t;
typedef __attribute__((ext_vector_type(16))) __bf16 v16bf;
typedef __attribute__((ext_vector_type(8)))  __bf16 v8bf;
typedef __attribute__((ext_vector_type(8)))  float  v8f;

static __device__ inline v8f wmma_bf16(v16bf a, v16bf b, v8f c) {
  // D = A(16x32 bf16) x B(32x16 bf16) + C(16x16 f32)
  return __builtin_amdgcn_wmma_f32_16x16x32_bf16(
      false, a, false, b, (short)0, c, false, false);
}

// A fragment: p0 = &A[row][k0 + 16*half]; 16 contiguous elems as two 8-elem (16B) chunks.
static __device__ inline v16bf load_a16x32(const bf16_t* p0) {
  union { v16bf v; v8bf h[2]; } u;
  u.h[0] = *(const v8bf*)(p0);
  u.h[1] = *(const v8bf*)(p0 + 8);
  return u.v;
}

// Async DMA: copy 16 bytes global -> LDS, tracked by ASYNCcnt (CDNA5 path).
// Per-lane addresses; lds_off is the wave-relative LDS byte address.
static __device__ inline void async_copy16(unsigned lds_off, const void* gptr) {
  asm volatile("global_load_async_to_lds_b128 %0, %1, off"
               :: "v"(lds_off), "v"((unsigned long long)(size_t)gptr)
               : "memory");
}
static __device__ inline unsigned lds_addr(const void* p) {
  return (unsigned)(size_t)p;  // low 32 bits of flat LDS address = LDS offset
}

// ---------------------------------------------------------------- weights
// in: [K][N] f32  ->  out: [N][K] bf16 (transposed, B-fragment friendly)
__global__ __launch_bounds__(256) void cvt_transpose_kernel(
    const float* __restrict__ in, bf16_t* __restrict__ out, int K, int N) {
  size_t i = (size_t)blockIdx.x * blockDim.x + threadIdx.x;
  size_t total = (size_t)K * N;
  if (i < total) {
    int n = (int)(i % N);
    int k = (int)(i / N);
    out[(size_t)n * K + k] = (bf16_t)in[i];
  }
}

// ---------------------------------------------------------------- layernorm
__global__ __launch_bounds__(256) void ln_kernel(
    const float* __restrict__ x, const float* __restrict__ g,
    const float* __restrict__ b, bf16_t* __restrict__ out) {
  int row = blockIdx.x;
  int tid = threadIdx.x;
  const float* xr = x + (size_t)row * CDIM;
  float v0 = xr[tid], v1 = xr[tid + 256], v2 = xr[tid + 512];
  float s  = v0 + v1 + v2;
  float s2 = v0 * v0 + v1 * v1 + v2 * v2;
#pragma unroll
  for (int off = 16; off >= 1; off >>= 1) {
    s  += __shfl_xor(s,  off, 32);
    s2 += __shfl_xor(s2, off, 32);
  }
  __shared__ float red[16];
  int wib = tid >> 5, lane = tid & 31;
  if (lane == 0) { red[wib] = s; red[8 + wib] = s2; }
  __syncthreads();
  float S = 0.f, S2 = 0.f;
#pragma unroll
  for (int i = 0; i < 8; ++i) { S += red[i]; S2 += red[8 + i]; }
  float mu   = S * (1.0f / CDIM);
  float var  = S2 * (1.0f / CDIM) - mu * mu;
  float rstd = rsqrtf(var + 1e-5f);
  bf16_t* orow = out + (size_t)row * CDIM;
  orow[tid]       = (bf16_t)((v0 - mu) * rstd * g[tid]       + b[tid]);
  orow[tid + 256] = (bf16_t)((v1 - mu) * rstd * g[tid + 256] + b[tid + 256]);
  orow[tid + 512] = (bf16_t)((v2 - mu) * rstd * g[tid + 512] + b[tid + 512]);
}

// ---------------------------------------------------------------- GEMM
// Block = 8 waves, tile 256(M) x 64(N). Wave tile 32x64 (8 accumulators).
// B (weights, transposed [N][K]) staged per-block in LDS via async DMA,
// double-buffered; branch-free steady-state loop with peeled tail.
// mode 1: QKV epilogue -> Q (x0.125, [h][tok][64]), K ([h][tok][64]),
//         Vt ([h][64][tok]) bf16
// mode 2: outF = acc + bias + res   (f32 residual stream)
// mode 3: outB = gelu(acc + bias)   (bf16, exact erf gelu)
__global__ __launch_bounds__(256) void gemm_wmma_kernel(
    const bf16_t* __restrict__ A, const bf16_t* __restrict__ Bt,
    const float* __restrict__ bias, const float* __restrict__ res,
    float* __restrict__ outF, bf16_t* __restrict__ outB,
    bf16_t* __restrict__ qb, bf16_t* __restrict__ kb_, bf16_t* __restrict__ vtb,
    int M, int N, int K, int mode) {
  __shared__ __align__(64) bf16_t ldsB[2][64][32];  // [n][k] per 32-k step
  const int tid  = threadIdx.x;
  const int lane = tid & 31;
  const int wib  = tid >> 5;
  const int tiles_n = N >> 6;
  const int mblk = blockIdx.x / tiles_n;
  const int nt   = blockIdx.x - mblk * tiles_n;
  const int n0   = nt << 6;
  const int row0 = mblk * 256 + wib * 32;
  const int m16 = lane & 15, hf = lane >> 4;

  // async-copy assignment: thread -> (bn = n row of Bt strip, bc = 16B chunk)
  const int bn = tid >> 2, bc = tid & 3;
  const bf16_t* bsrc = Bt + (size_t)(n0 + bn) * K + bc * 8;
  const unsigned bdst0 = lds_addr(&ldsB[0][bn][bc * 8]);
  const unsigned bdst1 = lds_addr(&ldsB[1][bn][bc * 8]);

  const bf16_t* ap0 = A + (size_t)(row0 + m16) * K + 16 * hf;
  const bf16_t* ap1 = A + (size_t)(row0 + 16 + m16) * K + 16 * hf;

  v8f acc[2][4] = {};

  // One k-step: hoist all 4 B fragments, then 8 back-to-back WMMAs.
  auto kstep = [&](const bf16_t (*bt)[32], int k) {
    v16bf a0 = load_a16x32(ap0 + k);
    v16bf a1 = load_a16x32(ap1 + k);
    v16bf b0 = *(const v16bf*)&bt[     m16][16 * hf];
    v16bf b1 = *(const v16bf*)&bt[16 + m16][16 * hf];
    v16bf b2 = *(const v16bf*)&bt[32 + m16][16 * hf];
    v16bf b3 = *(const v16bf*)&bt[48 + m16][16 * hf];
    acc[0][0] = wmma_bf16(a0, b0, acc[0][0]);
    acc[1][0] = wmma_bf16(a1, b0, acc[1][0]);
    acc[0][1] = wmma_bf16(a0, b1, acc[0][1]);
    acc[1][1] = wmma_bf16(a1, b1, acc[1][1]);
    acc[0][2] = wmma_bf16(a0, b2, acc[0][2]);
    acc[1][2] = wmma_bf16(a1, b2, acc[1][2]);
    acc[0][3] = wmma_bf16(a0, b3, acc[0][3]);
    acc[1][3] = wmma_bf16(a1, b3, acc[1][3]);
  };

  const int nsteps = K >> 5;
  async_copy16(bdst0, bsrc);  // prologue: stage k=0 into buf0
  for (int s = 0; s < nsteps - 1; ++s) {
    const int k = s << 5;
    async_copy16(((s + 1) & 1) ? bdst1 : bdst0, bsrc + k + 32);
    asm volatile("s_wait_asynccnt 1" ::: "memory");
    __syncthreads();  // all waves' copies for current buffer visible
    kstep(ldsB[s & 1], k);
    __syncthreads();  // done reading buffer; safe to overwrite next iter
  }
  {  // peeled tail
    asm volatile("s_wait_asynccnt 0" ::: "memory");
    __syncthreads();
    kstep(ldsB[(nsteps - 1) & 1], (nsteps - 1) << 5);
  }

#pragma unroll
  for (int sub = 0; sub < 2; ++sub) {
#pragma unroll
    for (int t = 0; t < 4; ++t) {
      const int col = n0 + t * 16 + m16;
      const float bv = bias[col];
#pragma unroll
      for (int r = 0; r < 8; ++r) {
        const int row = row0 + sub * 16 + r + 8 * hf;
        float val = acc[sub][t][r] + bv;
        if (mode == 1) {
          int gsel = col / CDIM;
          int cw   = col - gsel * CDIM;
          int head = cw >> 6;
          int d    = cw & 63;
          if (gsel == 0)
            qb[((size_t)(head * NTOK + row) << 6) + d] = (bf16_t)(val * 0.125f);
          else if (gsel == 1)
            kb_[((size_t)(head * NTOK + row) << 6) + d] = (bf16_t)val;
          else
            vtb[(size_t)(head * HDIM + d) * NTOK + row] = (bf16_t)val;
        } else if (mode == 2) {
          size_t idx = (size_t)row * N + col;
          outF[idx] = val + res[idx];
        } else {  // mode 3: exact GELU -> bf16
          float gel = 0.5f * val * (1.0f + erff(val * 0.70710678118654752f));
          outB[(size_t)row * N + col] = (bf16_t)gel;
        }
      }
    }
  }
}

// ---------------------------------------------------------------- attention
// Block = 8 waves of the SAME head (8 query tiles of 16). K/V chunks (32 keys)
// staged once per block in LDS via async DMA, double-buffered. Flash-style
// online softmax per wave. Q pre-scaled by 1/sqrt(64) at QKV epilogue.
__global__ __launch_bounds__(256) void attn_kernel(
    const bf16_t* __restrict__ Q, const bf16_t* __restrict__ Km,
    const bf16_t* __restrict__ Vt, bf16_t* __restrict__ O) {
  __shared__ __align__(64) bf16_t ldsK[2][32][64];  // [key][dim]
  __shared__ __align__(64) bf16_t ldsV[2][64][32];  // [dim][key]
  __shared__ __align__(64) bf16_t sp[8][16][32];    // per-wave P staging
  const int tid  = threadIdx.x;
  const int lane = tid & 31;
  const int wib  = tid >> 5;
  const int hh = blockIdx.x >> 5;                    // 32 blocks per head
  const int qt = ((blockIdx.x & 31) << 3) + wib;     // wave's 16-query tile
  const int m16 = lane & 15, hf = lane >> 4;

  // Q A-fragments (k-dim = 64 -> two fragments)
  const bf16_t* qp = Q + (size_t)(hh * NTOK + qt * 16 + m16) * HDIM + 16 * hf;
  v16bf aq0 = load_a16x32(qp);
  v16bf aq1 = load_a16x32(qp + 32);

  // async-copy assignments: K chunk 32x64 (8 chunks/row), V chunk 64x32 (4/row)
  const int kn = tid >> 3, kc = tid & 7;
  const bf16_t* ksrc = Km + (size_t)(hh * NTOK + kn) * HDIM + kc * 8;
  const unsigned kdst0 = lds_addr(&ldsK[0][kn][kc * 8]);
  const unsigned kdst1 = lds_addr(&ldsK[1][kn][kc * 8]);
  const int vd = tid >> 2, vc = tid & 3;
  const bf16_t* vsrc = Vt + (size_t)(hh * HDIM + vd) * NTOK + vc * 8;
  const unsigned vdst0 = lds_addr(&ldsV[0][vd][vc * 8]);
  const unsigned vdst1 = lds_addr(&ldsV[1][vd][vc * 8]);

  v8f o0 = {}, o1 = {}, o2 = {}, o3 = {};
  float mrow[8], lrow[8];
#pragma unroll
  for (int r = 0; r < 8; ++r) { mrow[r] = -3.0e38f; lrow[r] = 0.f; }

  // One 32-key chunk: S WMMAs, early V loads, online softmax, P reshape, PV.
  auto chunk = [&](const bf16_t (*tk)[64], const bf16_t (*tv)[32]) {
    v16bf bk00 = *(const v16bf*)&tk[     m16][     16 * hf];
    v16bf bk01 = *(const v16bf*)&tk[     m16][32 + 16 * hf];
    v16bf bk10 = *(const v16bf*)&tk[16 + m16][     16 * hf];
    v16bf bk11 = *(const v16bf*)&tk[16 + m16][32 + 16 * hf];
    v8f s0 = {}, s1 = {};
    s0 = wmma_bf16(aq0, bk00, s0);
    s1 = wmma_bf16(aq0, bk10, s1);
    s0 = wmma_bf16(aq1, bk01, s0);
    s1 = wmma_bf16(aq1, bk11, s1);
    // issue V fragment loads early; latency hides under softmax VALU work
    v16bf bv0 = *(const v16bf*)&tv[     m16][16 * hf];
    v16bf bv1 = *(const v16bf*)&tv[16 + m16][16 * hf];
    v16bf bv2 = *(const v16bf*)&tv[32 + m16][16 * hf];
    v16bf bv3 = *(const v16bf*)&tv[48 + m16][16 * hf];

    float p0[8], p1[8], cm[8];
#pragma unroll
    for (int r = 0; r < 8; ++r) cm[r] = fmaxf(s0[r], s1[r]);
#pragma unroll
    for (int off = 1; off < 16; off <<= 1) {
#pragma unroll
      for (int r = 0; r < 8; ++r) cm[r] = fmaxf(cm[r], __shfl_xor(cm[r], off, 16));
    }
    float alpha[8], rs[8];
#pragma unroll
    for (int r = 0; r < 8; ++r) {
      float nm = fmaxf(mrow[r], cm[r]);
      alpha[r] = __expf(mrow[r] - nm);
      mrow[r] = nm;
      float e0 = __expf(s0[r] - nm);
      float e1 = __expf(s1[r] - nm);
      p0[r] = e0; p1[r] = e1; rs[r] = e0 + e1;
    }
#pragma unroll
    for (int off = 1; off < 16; off <<= 1) {
#pragma unroll
      for (int r = 0; r < 8; ++r) rs[r] += __shfl_xor(rs[r], off, 16);
    }
#pragma unroll
    for (int r = 0; r < 8; ++r) {
      lrow[r] = lrow[r] * alpha[r] + rs[r];
      o0[r] *= alpha[r]; o1[r] *= alpha[r];
      o2[r] *= alpha[r]; o3[r] *= alpha[r];
    }
    // P: D-layout (lane=key, reg=query) -> per-wave LDS [query][key]
#pragma unroll
    for (int r = 0; r < 8; ++r) {
      int rr = r + 8 * hf;
      sp[wib][rr][m16]      = (bf16_t)p0[r];
      sp[wib][rr][16 + m16] = (bf16_t)p1[r];
    }
    asm volatile("s_wait_dscnt 0" ::: "memory");
    union { v16bf v; v8bf h[2]; } up;  // P A-fragment from LDS
    up.h[0] = *(const v8bf*)&sp[wib][m16][16 * hf];
    up.h[1] = *(const v8bf*)&sp[wib][m16][16 * hf + 8];

    o0 = wmma_bf16(up.v, bv0, o0);
    o1 = wmma_bf16(up.v, bv1, o1);
    o2 = wmma_bf16(up.v, bv2, o2);
    o3 = wmma_bf16(up.v, bv3, o3);
  };

  const int nchunks = NTOK / 32;
  async_copy16(kdst0, ksrc);
  async_copy16(vdst0, vsrc);
  for (int s = 0; s < nchunks - 1; ++s) {
    const int kb = s << 5;
    async_copy16(((s + 1) & 1) ? kdst1 : kdst0, ksrc + (size_t)(kb + 32) * HDIM);
    async_copy16(((s + 1) & 1) ? vdst1 : vdst0, vsrc + kb + 32);
    asm volatile("s_wait_asynccnt 2" ::: "memory");
    __syncthreads();
    chunk(ldsK[s & 1], ldsV[s & 1]);
    __syncthreads();  // done reading K/V buffer before it is overwritten
  }
  {  // peeled tail
    asm volatile("s_wait_asynccnt 0" ::: "memory");
    __syncthreads();
    chunk(ldsK[(nchunks - 1) & 1], ldsV[(nchunks - 1) & 1]);
  }

#pragma unroll
  for (int r = 0; r < 8; ++r) {
    float inv = 1.0f / lrow[r];
    int tok = qt * 16 + r + 8 * hf;
    bf16_t* op = O + (size_t)tok * CDIM + hh * HDIM + m16;
    op[0]  = (bf16_t)(o0[r] * inv);
    op[16] = (bf16_t)(o1[r] * inv);
    op[32] = (bf16_t)(o2[r] * inv);
    op[48] = (bf16_t)(o3[r] * inv);
  }
}

// ---------------------------------------------------------------- launch
extern "C" void kernel_launch(void* const* d_in, const int* in_sizes, int n_in,
                              void* d_out, int out_size, void* d_ws, size_t ws_size,
                              hipStream_t stream) {
  const float* x      = (const float*)d_in[0];
  const float* ln1_g  = (const float*)d_in[1];
  const float* ln1_b  = (const float*)d_in[2];
  const float* qkv_w  = (const float*)d_in[3];
  const float* qkv_b  = (const float*)d_in[4];
  const float* proj_w = (const float*)d_in[5];
  const float* proj_b = (const float*)d_in[6];
  const float* ln2_g  = (const float*)d_in[7];
  const float* ln2_b  = (const float*)d_in[8];
  const float* fc1_w  = (const float*)d_in[9];
  const float* fc1_b  = (const float*)d_in[10];
  const float* fc2_w  = (const float*)d_in[11];
  const float* fc2_b  = (const float*)d_in[12];
  float* out = (float*)d_out;

  char* ws = (char*)d_ws;
  size_t off = 0;
  auto balloc = [&](size_t elems) -> bf16_t* {
    bf16_t* p = (bf16_t*)(ws + off);
    off = (off + elems * sizeof(bf16_t) + 255) & ~(size_t)255;
    return p;
  };
  auto falloc = [&](size_t elems) -> float* {
    float* p = (float*)(ws + off);
    off = (off + elems * sizeof(float) + 255) & ~(size_t)255;
    return p;
  };

  bf16_t* qkvw_t  = balloc((size_t)(3 * CDIM) * CDIM);  // [2304][768]
  bf16_t* projw_t = balloc((size_t)CDIM * CDIM);        // [768][768]
  bf16_t* fc1w_t  = balloc((size_t)HID * CDIM);         // [3072][768]
  bf16_t* fc2w_t  = balloc((size_t)CDIM * HID);         // [768][3072]
  bf16_t* h1  = balloc((size_t)NTOK * CDIM);
  bf16_t* Qb  = balloc((size_t)NHEAD * NTOK * HDIM);
  bf16_t* Kb  = balloc((size_t)NHEAD * NTOK * HDIM);
  bf16_t* Vt  = balloc((size_t)NHEAD * HDIM * NTOK);
  bf16_t* att = balloc((size_t)NTOK * CDIM);
  float*  x1  = falloc((size_t)NTOK * CDIM);
  bf16_t* h2  = balloc((size_t)NTOK * CDIM);
  bf16_t* h3  = balloc((size_t)NTOK * HID);

  // --- weight convert + transpose (one-shot, L2-resident afterwards)
  cvt_transpose_kernel<<<(CDIM * 3 * CDIM + 255) / 256, 256, 0, stream>>>(
      qkv_w, qkvw_t, CDIM, 3 * CDIM);
  cvt_transpose_kernel<<<(CDIM * CDIM + 255) / 256, 256, 0, stream>>>(
      proj_w, projw_t, CDIM, CDIM);
  cvt_transpose_kernel<<<(CDIM * HID + 255) / 256, 256, 0, stream>>>(
      fc1_w, fc1w_t, CDIM, HID);
  cvt_transpose_kernel<<<(HID * CDIM + 255) / 256, 256, 0, stream>>>(
      fc2_w, fc2w_t, HID, CDIM);

  // --- attention branch
  ln_kernel<<<NTOK, 256, 0, stream>>>(x, ln1_g, ln1_b, h1);
  gemm_wmma_kernel<<<(NTOK / 256) * ((3 * CDIM) / 64), 256, 0, stream>>>(
      h1, qkvw_t, qkv_b, nullptr, nullptr, nullptr,
      Qb, Kb, Vt, NTOK, 3 * CDIM, CDIM, 1);
  attn_kernel<<<NHEAD * (NTOK / 16) / 8, 256, 0, stream>>>(Qb, Kb, Vt, att);
  gemm_wmma_kernel<<<(NTOK / 256) * (CDIM / 64), 256, 0, stream>>>(
      att, projw_t, proj_b, x, x1, nullptr,
      nullptr, nullptr, nullptr, NTOK, CDIM, CDIM, 2);

  // --- MLP branch
  ln_kernel<<<NTOK, 256, 0, stream>>>(x1, ln2_g, ln2_b, h2);
  gemm_wmma_kernel<<<(NTOK / 256) * (HID / 64), 256, 0, stream>>>(
      h2, fc1w_t, fc1_b, nullptr, nullptr, h3,
      nullptr, nullptr, nullptr, NTOK, HID, CDIM, 3);
  gemm_wmma_kernel<<<(NTOK / 256) * (CDIM / 64), 256, 0, stream>>>(
      h3, fc2w_t, fc2_b, x1, out, nullptr,
      nullptr, nullptr, nullptr, NTOK, CDIM, HID, 2);
}
